// SpatialSelfAttention_58884001628507
// MI455X (gfx1250) — matrix operand order repaired and checked by reading, not compiled, not run on hardware
//
#include <hip/hip_runtime.h>
#include <hip/hip_fp16.h>

typedef __attribute__((ext_vector_type(16))) _Float16 v16h;
typedef __attribute__((ext_vector_type(8)))  _Float16 v8h;
typedef __attribute__((ext_vector_type(8)))  float    v8f;

#define CH    1024
#define NH    16
#define HD    64
#define BATCH 8
#define SEQ   1024
#define TOK   (BATCH * SEQ)   // 8192 tokens

// ---------------------------------------------------------------------------
// WMMA fragment helpers (wave32, v_wmma_f32_16x16x32_f16 layouts, ISA 7.12.2)
// ---------------------------------------------------------------------------

// A (16x32, f16) from row-major [M][K] tile: lane holds row (lane&15),
// halves 0..7 -> K = kb..kb+7, halves 8..15 -> K = kb+16..kb+23, kb=(lane>>4)*8
__device__ __forceinline__ v16h load_a_frag(const _Float16* tile, int ld, int lane) {
    const int m  = lane & 15;
    const int kb = (lane >> 4) * 8;
    const _Float16* p = tile + m * ld + kb;
    v8h lo = *(const v8h*)(p);
    v8h hi = *(const v8h*)(p + 16);
    return __builtin_shufflevector(lo, hi, 0,1,2,3,4,5,6,7,8,9,10,11,12,13,14,15);
}

// B (32x16, f16) from B^T stored row-major [N][K]: lane holds column (lane&15),
// halves 0..7 -> K = kb..kb+7, halves 8..15 -> K = kb+16..kb+23, kb=(lane>>4)*8
__device__ __forceinline__ v16h load_bT_frag(const _Float16* tile, int ld, int lane) {
    const int n  = lane & 15;
    const int kb = (lane >> 4) * 8;
    const _Float16* p = tile + n * ld + kb;
    v8h lo = *(const v8h*)(p);
    v8h hi = *(const v8h*)(p + 16);
    return __builtin_shufflevector(lo, hi, 0,1,2,3,4,5,6,7,8,9,10,11,12,13,14,15);
}

__device__ __forceinline__ v8f wmma16(v16h a, v16h b, v8f c) {
    return __builtin_amdgcn_wmma_f32_16x16x32_f16(false, a, false, b,
                                                  (short)0, c, false, false);
}

// ---------------------------------------------------------------------------
// CDNA5 async global->LDS copy (ASYNCcnt path).  16 bytes per lane.
// lds_off is the byte offset within the workgroup's LDS (dynamic LDS starts
// at 0 here: no static __shared__ in these kernels).
// ---------------------------------------------------------------------------
__device__ __forceinline__ void async_copy_b128(unsigned lds_off, const _Float16* g) {
    unsigned long long ga = (unsigned long long)(__SIZE_TYPE__)g;
    asm volatile("global_load_async_to_lds_b128 %0, %1, off"
                 :: "v"(lds_off), "v"(ga) : "memory");
}
__device__ __forceinline__ void wait_async_le8()  {
    asm volatile("s_wait_asynccnt 8" ::: "memory");
}
__device__ __forceinline__ void wait_async_zero() {
    asm volatile("s_wait_asynccnt 0" ::: "memory");
}

// Cooperative async stage of a 128-row x 64-half tile (16 KB) from row-major
// global (leading dim ldg halves) into LDS at byte offset lds_base.
// 256 threads: 2 threads/row, 4 x b128 (64B) each -> 4 async ops per thread.
__device__ __forceinline__ void stage_tile_async(const _Float16* gbase, int ldg,
                                                 unsigned lds_base, int tid) {
    const int row  = tid >> 1;
    const int colh = (tid & 1) * 32;                       // halves
    const _Float16* g = gbase + (__SIZE_TYPE__)row * ldg + colh;
    const unsigned l  = lds_base + (unsigned)(row * 64 + colh) * 2;
#pragma unroll
    for (int i = 0; i < 4; ++i)
        async_copy_b128(l + i * 16, g + i * 8);
}

// LDS byte offsets for double-buffered 128x64 A and B^T tiles (64 KB total)
#define LDS_A0 0u
#define LDS_B0 16384u
#define LDS_A1 32768u
#define LDS_B1 49152u

// ---------------------------------------------------------------------------
// Prep kernels: fp32 -> f16, and weight transposes to [out][in]
// ---------------------------------------------------------------------------

__global__ void __launch_bounds__(256) cvt_f32_to_f16(const float* __restrict__ in,
                                                      _Float16* __restrict__ out, int n) {
    int i = blockIdx.x * blockDim.x + threadIdx.x;
    if (i < n) out[i] = (_Float16)in[i];
}

// in: [rows_in][cols_in] f32 ; out: [cols_in][rows_in] f16 (row-major)
__global__ void __launch_bounds__(256) transpose_to_f16(const float* __restrict__ in,
                                                        _Float16* __restrict__ out,
                                                        int rows_in, int cols_in) {
    int idx = blockIdx.x * blockDim.x + threadIdx.x;
    if (idx >= rows_in * cols_in) return;
    int c = idx / rows_in;   // output row  (= input column)
    int r = idx % rows_in;   // output col  (= input row)
    out[idx] = (_Float16)in[r * cols_in + c];
}

// ---------------------------------------------------------------------------
// GEMM 1: qkv = x @ W_qkv  (M=8192, N=3072, K=1024), scatter to head layouts.
// Workgroup computes a 128(M) x 128(N) block; 8 waves as 4(M) x 2(N) grid of
// 32x64 wave tiles.  K staged in 64-wide chunks, double-buffered via
// global_load_async_to_lds_b128 (ASYNCcnt).
//   q,k : [B][H][N][D] f16   (k row-major serves as B-frag source for q.k^T)
//   vT  : [B][H][D][N] f16   (so PV GEMM B-frag K-dim = key idx is contiguous)
// ---------------------------------------------------------------------------
__global__ void __launch_bounds__(256) qkv_gemm(const _Float16* __restrict__ xh,
                                                const _Float16* __restrict__ WqkvT,
                                                _Float16* __restrict__ qh,
                                                _Float16* __restrict__ kh,
                                                _Float16* __restrict__ vTh) {
    extern __shared__ char smem[];
    const int tid  = threadIdx.x;
    const int lane = tid & 31;
    const int w    = tid >> 5;         // 0..7
    const int wm   = w & 3;            // 4 M sub-tiles of 32
    const int wn   = w >> 2;           // 2 N sub-tiles of 64
    const int nb   = blockIdx.x;       // 0..23  (128-col blocks of 3072)
    const int mb   = blockIdx.y;       // 0..63  (128-row blocks of 8192)

    const _Float16* Ag = xh    + (__SIZE_TYPE__)(mb * 128) * CH;
    const _Float16* Bg = WqkvT + (__SIZE_TYPE__)(nb * 128) * CH;

    const unsigned bufA[2] = {LDS_A0, LDS_A1};
    const unsigned bufB[2] = {LDS_B0, LDS_B1};

    // Prefetch chunk 0
    stage_tile_async(Ag, CH, bufA[0], tid);
    stage_tile_async(Bg, CH, bufB[0], tid);

    v8f acc[2][4] = {};
    const int NKC = CH / 64;           // 16 chunks
    for (int kc = 0; kc < NKC; ++kc) {
        const int cur = kc & 1;
        if (kc + 1 < NKC) {
            stage_tile_async(Ag + (kc + 1) * 64, CH, bufA[cur ^ 1], tid);
            stage_tile_async(Bg + (kc + 1) * 64, CH, bufB[cur ^ 1], tid);
            wait_async_le8();          // chunk kc's 8 ops complete (in-order)
        } else {
            wait_async_zero();
        }
        __syncthreads();               // whole tile resident

        const _Float16* Ab = (const _Float16*)(smem + bufA[cur]);
        const _Float16* Bb = (const _Float16*)(smem + bufB[cur]);
#pragma unroll
        for (int ks = 0; ks < 2; ++ks) {
            v16h a0 = load_a_frag(Ab + (wm * 32)      * 64 + ks * 32, 64, lane);
            v16h a1 = load_a_frag(Ab + (wm * 32 + 16) * 64 + ks * 32, 64, lane);
#pragma unroll
            for (int t = 0; t < 4; ++t) {
                v16h bf = load_bT_frag(Bb + (wn * 64 + t * 16) * 64 + ks * 32, 64, lane);
                acc[0][t] = wmma16(a0, bf, acc[0][t]);
                acc[1][t] = wmma16(a1, bf, acc[1][t]);
            }
        }
        __syncthreads();               // done reading cur before it's re-staged
    }

    const int nlo = lane & 15;
    const int mhi = (lane >> 4) * 8;
#pragma unroll
    for (int mi = 0; mi < 2; ++mi) {
#pragma unroll
        for (int t = 0; t < 4; ++t) {
            const int c     = nb * 128 + wn * 64 + t * 16 + nlo;   // 0..3071
            const int which = c >> 10;                  // 0=q 1=k 2=v (uniform/tile)
            const int cc    = c & (CH - 1);
            const int h     = cc >> 6;
            const int d     = cc & (HD - 1);
#pragma unroll
            for (int r = 0; r < 8; ++r) {
                const int tok = mb * 128 + wm * 32 + mi * 16 + mhi + r;
                const int b   = tok >> 10;
                const int n   = tok & (SEQ - 1);
                const _Float16 val = (_Float16)acc[mi][t][r];
                if (which == 0)
                    qh[(__SIZE_TYPE__)((b * NH + h) * SEQ + n) * HD + d] = val;
                else if (which == 1)
                    kh[(__SIZE_TYPE__)((b * NH + h) * SEQ + n) * HD + d] = val;
                else
                    vTh[(__SIZE_TYPE__)((b * NH + h) * HD + d) * SEQ + n] = val;
            }
        }
    }
}

// ---------------------------------------------------------------------------
// Attention: one block (2 waves, 64 threads) per (b, h, 32-query tile).
// LDS: S f32[32][1024] (128KB) + P f16[32][1024] (64KB) + rowsum f32[32].
// Pass1: S = q.k^T * scale (WMMA). Pass2: softmax numerator into P + row sums.
// Pass3: out = (P @ v) / rowsum (WMMA, A-frags from LDS), f16 [tok][C].
// ---------------------------------------------------------------------------
__global__ void __launch_bounds__(64) attention(const _Float16* __restrict__ qh,
                                                const _Float16* __restrict__ kh,
                                                const _Float16* __restrict__ vTh,
                                                _Float16* __restrict__ attn_out) {
    extern __shared__ char smem[];
    float*    S    = (float*)smem;                          // 32*1024 f32
    _Float16* P    = (_Float16*)(smem + 32 * SEQ * 4);      // 32*1024 f16
    float*    rsum = (float*)(smem + 32 * SEQ * 6);         // 32 f32

    const int lane = threadIdx.x & 31;
    const int w    = threadIdx.x >> 5;          // wave in block: 0..1
    const int qb   = blockIdx.x;                // 0..31  (32-query tiles)
    const int h    = blockIdx.y;
    const int b    = blockIdx.z;

    const _Float16* qhead = qh  + (__SIZE_TYPE__)((b * NH + h) * SEQ) * HD;
    const _Float16* khead = kh  + (__SIZE_TYPE__)((b * NH + h) * SEQ) * HD;
    const _Float16* vhead = vTh + (__SIZE_TYPE__)((b * NH + h) * HD) * SEQ;

    const int q0  = qb * 32 + w * 16;
    const int nlo = lane & 15;
    const int mhi = (lane >> 4) * 8;

    // ---- Pass 1: scores -------------------------------------------------
    v16h a0 = load_a_frag(qhead + (__SIZE_TYPE__)q0 * HD,      HD, lane);  // d 0..31
    v16h a1 = load_a_frag(qhead + (__SIZE_TYPE__)q0 * HD + 32, HD, lane);  // d 32..63
    for (int nt = 0; nt < SEQ / 16; ++nt) {
        v8f acc = {};
        v16h b0 = load_bT_frag(khead + (__SIZE_TYPE__)(nt * 16) * HD,      HD, lane);
        v16h b1 = load_bT_frag(khead + (__SIZE_TYPE__)(nt * 16) * HD + 32, HD, lane);
        acc = wmma16(a0, b0, acc);
        acc = wmma16(a1, b1, acc);
        float* srow = S + (w * 16) * SEQ + nt * 16 + nlo;
#pragma unroll
        for (int r = 0; r < 8; ++r)
            srow[(mhi + r) * SEQ] = acc[r] * 0.125f;   // 1/sqrt(64)
    }
    __syncthreads();

    // ---- Pass 2: softmax numerator (2 lanes per row) --------------------
    {
        const int row = w * 16 + (lane & 15);
        const int c0  = (lane >> 4) * (SEQ / 2);
        const float* sr = S + row * SEQ + c0;
        float m = -1e30f;
        for (int i = 0; i < SEQ / 2; ++i) m = fmaxf(m, sr[i]);
        m = fmaxf(m, __shfl_xor(m, 16, 32));
        float sum = 0.f;
        _Float16* pr = P + row * SEQ + c0;
        for (int i = 0; i < SEQ / 2; ++i) {
            float e = __expf(sr[i] - m);
            sum += e;
            pr[i] = (_Float16)e;
        }
        sum += __shfl_xor(sum, 16, 32);
        if (lane < 16) rsum[row] = sum;
    }
    __syncthreads();

    // ---- Pass 3: out = (P @ v) / rowsum ---------------------------------
    const _Float16* Pw = P + (w * 16) * SEQ;
    v8f acc[4] = {};
    for (int ks = 0; ks < SEQ / 32; ++ks) {
        v16h a = load_a_frag(Pw + ks * 32, SEQ, lane);
#pragma unroll
        for (int t = 0; t < 4; ++t) {
            v16h bf = load_bT_frag(vhead + (__SIZE_TYPE__)(t * 16) * SEQ + ks * 32,
                                   SEQ, lane);
            acc[t] = wmma16(a, bf, acc[t]);
        }
    }
#pragma unroll
    for (int t = 0; t < 4; ++t) {
#pragma unroll
        for (int r = 0; r < 8; ++r) {
            const int row = mhi + r;                       // 0..15 in wave tile
            const float inv = 1.0f / rsum[w * 16 + row];
            const int tok = b * SEQ + q0 + row;
            const int c   = h * HD + t * 16 + nlo;
            attn_out[(__SIZE_TYPE__)tok * CH + c] = (_Float16)(acc[t][r] * inv);
        }
    }
}

// ---------------------------------------------------------------------------
// GEMM 3: out = attn_out @ W_out  (M=8192, N=1024, K=1024), f32 result.
// Same 128x128 LDS-tiled async-staged structure as qkv_gemm.
// ---------------------------------------------------------------------------
__global__ void __launch_bounds__(256) out_gemm(const _Float16* __restrict__ ah,
                                                const _Float16* __restrict__ WoutT,
                                                float* __restrict__ out) {
    extern __shared__ char smem[];
    const int tid  = threadIdx.x;
    const int lane = tid & 31;
    const int w    = tid >> 5;
    const int wm   = w & 3;
    const int wn   = w >> 2;
    const int nb   = blockIdx.x;       // 0..7
    const int mb   = blockIdx.y;       // 0..63

    const _Float16* Ag = ah    + (__SIZE_TYPE__)(mb * 128) * CH;
    const _Float16* Bg = WoutT + (__SIZE_TYPE__)(nb * 128) * CH;

    const unsigned bufA[2] = {LDS_A0, LDS_A1};
    const unsigned bufB[2] = {LDS_B0, LDS_B1};

    stage_tile_async(Ag, CH, bufA[0], tid);
    stage_tile_async(Bg, CH, bufB[0], tid);

    v8f acc[2][4] = {};
    const int NKC = CH / 64;
    for (int kc = 0; kc < NKC; ++kc) {
        const int cur = kc & 1;
        if (kc + 1 < NKC) {
            stage_tile_async(Ag + (kc + 1) * 64, CH, bufA[cur ^ 1], tid);
            stage_tile_async(Bg + (kc + 1) * 64, CH, bufB[cur ^ 1], tid);
            wait_async_le8();
        } else {
            wait_async_zero();
        }
        __syncthreads();

        const _Float16* Ab = (const _Float16*)(smem + bufA[cur]);
        const _Float16* Bb = (const _Float16*)(smem + bufB[cur]);
#pragma unroll
        for (int ks = 0; ks < 2; ++ks) {
            v16h a0 = load_a_frag(Ab + (wm * 32)      * 64 + ks * 32, 64, lane);
            v16h a1 = load_a_frag(Ab + (wm * 32 + 16) * 64 + ks * 32, 64, lane);
#pragma unroll
            for (int t = 0; t < 4; ++t) {
                v16h bf = load_bT_frag(Bb + (wn * 64 + t * 16) * 64 + ks * 32, 64, lane);
                acc[0][t] = wmma16(a0, bf, acc[0][t]);
                acc[1][t] = wmma16(a1, bf, acc[1][t]);
            }
        }
        __syncthreads();
    }

    const int nlo = lane & 15;
    const int mhi = (lane >> 4) * 8;
#pragma unroll
    for (int mi = 0; mi < 2; ++mi) {
#pragma unroll
        for (int t = 0; t < 4; ++t) {
            const int c = nb * 128 + wn * 64 + t * 16 + nlo;
#pragma unroll
            for (int r = 0; r < 8; ++r) {
                const int tok = mb * 128 + wm * 32 + mi * 16 + mhi + r;
                out[(__SIZE_TYPE__)tok * CH + c] = acc[mi][t][r];
            }
        }
    }
}

// ---------------------------------------------------------------------------
// Host launch
// ---------------------------------------------------------------------------
extern "C" void kernel_launch(void* const* d_in, const int* in_sizes, int n_in,
                              void* d_out, int out_size, void* d_ws, size_t ws_size,
                              hipStream_t stream) {
    const float* x    = (const float*)d_in[0];   // [8,1024,1024]
    const float* Wqkv = (const float*)d_in[1];   // [1024,3072]
    const float* Wout = (const float*)d_in[2];   // [1024,1024]
    float* out = (float*)d_out;                  // [8,1024,1024]

    char* ws = (char*)d_ws;
    _Float16* xh    = (_Float16*)ws;                         // TOK*CH
    _Float16* WqkvT = xh    + (size_t)TOK * CH;              // 3C*C
    _Float16* WoutT = WqkvT + (size_t)3 * CH * CH;           // C*C
    _Float16* qh    = WoutT + (size_t)CH * CH;               // B*H*SEQ*HD == TOK*CH
    _Float16* kh    = qh    + (size_t)TOK * CH;
    _Float16* vTh   = kh    + (size_t)TOK * CH;
    _Float16* ah    = vTh   + (size_t)TOK * CH;              // attn output [tok][C]

    // Prep
    cvt_f32_to_f16<<<(TOK * CH) / 256, 256, 0, stream>>>(x, xh, TOK * CH);
    transpose_to_f16<<<(3 * CH * CH) / 256, 256, 0, stream>>>(Wqkv, WqkvT, CH, 3 * CH);
    transpose_to_f16<<<(CH * CH) / 256, 256, 0, stream>>>(Wout, WoutT, CH, CH);

    // QKV projection: 64 M-blocks x 24 N-blocks of 128x128, 64 KB LDS each
    qkv_gemm<<<dim3(24, 64), 256, 65536, stream>>>(xh, WqkvT, qh, kh, vTh);

    // Attention: (32 q-tiles, 16 heads, 8 batch), 2 waves, 192.1 KB LDS
    const size_t attn_lds = (size_t)32 * SEQ * 6 + 32 * 4;
    attention<<<dim3(32, NH, BATCH), 64, attn_lds, stream>>>(qh, kh, vTh, ah);

    // Output projection: 64 M-blocks x 8 N-blocks of 128x128
    out_gemm<<<dim3(8, 64), 256, 65536, stream>>>(ah, WoutT, out);
}
